// FlexMultiheadAttention_23733989278388
// MI455X (gfx1250) — matrix operand order repaired and checked by reading, not compiled, not run on hardware
//
#include <hip/hip_runtime.h>

// ---------------------------------------------------------------------------
// FlexMultiheadAttention forward for gfx1250 (MI455X), bf16 WMMA pipeline.
// B=4, S=1024, E=1024, H=16, D=64.
// d_out = [output (B*S*E) fp32 | attn_weights (B*S*S) fp32]
// ---------------------------------------------------------------------------

typedef __attribute__((ext_vector_type(16))) __bf16 v16bf;
typedef __attribute__((ext_vector_type(8)))  __bf16 v8bf;
typedef __attribute__((ext_vector_type(8)))  float  v8f;
typedef __attribute__((ext_vector_type(4)))  float  v4f;

#define NEG_INF (-__builtin_inff())

static __device__ __forceinline__ __bf16 f2bf(float f) {
  return (__bf16)f;   // native RNE convert (v_cvt_*bf16* on gfx1250)
}

static __device__ __forceinline__ v16bf cat8(v8bf lo, v8bf hi) {
  v16bf r;
#pragma unroll
  for (int i = 0; i < 8; ++i) { r[i] = lo[i]; r[i + 8] = hi[i]; }
  return r;
}

static __device__ __forceinline__ v8bf ld8(const __bf16* p) {
  return *(const v8bf*)p;
}

static __device__ __forceinline__ v8f wmma_bf16(v16bf a, v16bf b, v8f c) {
  return __builtin_amdgcn_wmma_f32_16x16x32_bf16(
      false, a, false, b, (short)0, c, false, false);
}

// half-wave (16-lane) reductions; XOR masks 1..8 stay within a 16-lane half.
static __device__ __forceinline__ float halfmax(float v) {
#pragma unroll
  for (int m = 1; m < 16; m <<= 1) v = fmaxf(v, __shfl_xor(v, m, 32));
  return v;
}
static __device__ __forceinline__ float halfsum(float v) {
#pragma unroll
  for (int m = 1; m < 16; m <<= 1) v += __shfl_xor(v, m, 32);
  return v;
}

// ---------------------------------------------------------------------------
// Kernel 0: one-shot fp32 -> bf16 conversion (8 elements / thread).
// ---------------------------------------------------------------------------
__global__ void fmha_cvt_bf16(const float* __restrict__ src,
                              __bf16* __restrict__ dst) {
  const size_t i = ((size_t)blockIdx.x * blockDim.x + threadIdx.x) * 8;
  v4f f0 = *(const v4f*)(src + i);
  v4f f1 = *(const v4f*)(src + i + 4);
  v8bf r;
#pragma unroll
  for (int j = 0; j < 4; ++j) { r[j] = f2bf(f0[j]); r[j + 4] = f2bf(f1[j]); }
  *(v8bf*)(dst + i) = r;
}

// ---------------------------------------------------------------------------
// Kernel 1: QKV projection.  C = X * W_m^T + b_m  (all-bf16 operands)
// One wave computes a 16x128 output tile.  blockIdx.y = m in {q,k,v}.
// q,k stored row-major bf16 [B*S][E]; v stored transposed per head:
// vtb[((b*16+h)*64+d)*1024 + s].
// ---------------------------------------------------------------------------
__global__ void fmha_qkv_proj(const __bf16* __restrict__ xq,
                              const __bf16* __restrict__ xk,
                              const __bf16* __restrict__ xv,
                              const __bf16* __restrict__ wib,
                              const float* __restrict__ b_in,
                              __bf16* __restrict__ qb,
                              __bf16* __restrict__ kb,
                              __bf16* __restrict__ vtb) {
  const int m    = blockIdx.y;
  const int wid  = threadIdx.x >> 5;
  const int lane = threadIdx.x & 31;
  const int hi   = lane >> 4;
  const int nl   = lane & 15;
  const int tile = blockIdx.x * 8 + wid;     // 0..2047
  const int rt   = tile >> 3;                // 256 row tiles of 16
  const int ct   = tile & 7;                 // 8 col tiles of 128
  const int row  = rt * 16 + nl;

  const __bf16* X  = (m == 0) ? xq : ((m == 1) ? xk : xv);
  const __bf16* Wm = wib + (size_t)m * 1024 * 1024;

  v8f acc[8] = {};
  for (int k0 = 0; k0 < 1024; k0 += 32) {
    const __bf16* ap = X + (size_t)row * 1024 + k0;
    v16bf a = cat8(ld8(ap + 16 * hi), ld8(ap + 8 + 16 * hi));
#pragma unroll
    for (int t = 0; t < 8; ++t) {
      const int n = ct * 128 + t * 16 + nl;
      const __bf16* bp = Wm + (size_t)n * 1024 + k0 + 16 * hi;
      acc[t] = wmma_bf16(a, cat8(ld8(bp), ld8(bp + 8)), acc[t]);
    }
  }

#pragma unroll
  for (int t = 0; t < 8; ++t) {
    const int col = ct * 128 + t * 16 + nl;
    const float bias = b_in[m * 1024 + col];
#pragma unroll
    for (int r = 0; r < 8; ++r) {
      const int grow = rt * 16 + r + 8 * hi;
      const float val = acc[t][r] + bias;
      if (m == 0) {
        qb[(size_t)grow * 1024 + col] = f2bf(val);
      } else if (m == 1) {
        kb[(size_t)grow * 1024 + col] = f2bf(val);
      } else {
        const int bb = grow >> 10, s = grow & 1023;
        const int h = col >> 6, d = col & 63;
        vtb[(((size_t)bb * 16 + h) * 64 + d) * 1024 + s] = f2bf(val);
      }
    }
  }
}

// score tile: 16 q-rows x 16 keys, K-dim = D = 64 (two bf16 WMMAs)
static __device__ __forceinline__ v8f score_tile(const __bf16* kptr, int hi,
                                                 v16bf a0, v16bf a1) {
  v16bf b0 = cat8(ld8(kptr + 16 * hi), ld8(kptr + 16 * hi + 8));
  v16bf b1 = cat8(ld8(kptr + 32 + 16 * hi), ld8(kptr + 40 + 16 * hi));
  v8f c = {};
  c = wmma_bf16(a0, b0, c);
  c = wmma_bf16(a1, b1, c);
  return c;
}

// ---------------------------------------------------------------------------
// Kernel 2: fused attention.  Grid (qblock=64, b=4), block = 512 = 16 waves,
// one wave per head.  Two-pass flash softmax; head-mean accumulated in LDS
// with ds_add_f32; ctx written bf16 for the out-proj GEMM.
// Dynamic LDS: 64KB mean buffer + 16 * 1KB P-staging = 80KB (<320KB/WGP).
// ---------------------------------------------------------------------------
__global__ void fmha_attention(const __bf16* __restrict__ qb,
                               const __bf16* __restrict__ kb,
                               const __bf16* __restrict__ vtb,
                               const unsigned char* __restrict__ kpm,
                               __bf16* __restrict__ ctxb,
                               float* __restrict__ attn_out) {
  extern __shared__ char smem[];
  float*  smean = (float*)smem;                                 // [16][1024]
  const int h    = threadIdx.x >> 5;                            // head
  const int lane = threadIdx.x & 31;
  const int hi   = lane >> 4;
  const int nl   = lane & 15;
  const int b    = blockIdx.y;
  const int qb0  = blockIdx.x * 16;
  __bf16* stg = (__bf16*)(smem + 65536) + h * (16 * 32);        // per-wave

  for (int i = threadIdx.x; i < 16 * 1024; i += 512) smean[i] = 0.0f;
  __syncthreads();

  // Q fragments for this wave's 16 rows, head h (K = D = 64 -> a0,a1)
  const __bf16* qp = qb + (size_t)((b << 10) + qb0 + nl) * 1024 + (h << 6);
  const v16bf a0 = cat8(ld8(qp + 16 * hi), ld8(qp + 8 + 16 * hi));
  const v16bf a1 = cat8(ld8(qp + 32 + 16 * hi), ld8(qp + 40 + 16 * hi));

  float m_[8], l_[8];
#pragma unroll
  for (int r = 0; r < 8; ++r) { m_[r] = NEG_INF; l_[r] = 0.0f; }

  // ---- pass 1: online row max / sum over all 64 key blocks ----
  for (int kc = 0; kc < 64; ++kc) {
    const __bf16* kptr =
        kb + (size_t)((b << 10) + (kc << 4) + nl) * 1024 + (h << 6);
    if (kc < 63) __builtin_prefetch(kptr + 1024, 0, 3);
    const v8f c = score_tile(kptr, hi, a0, a1);
    const bool keep = kpm[(b << 10) + (kc << 4) + nl] != 0;
    const int kcol = (kc << 4) + nl;
#pragma unroll
    for (int r = 0; r < 8; ++r) {
      const float s = keep
          ? (c[r] * 0.125f + (float)((qb0 + r + 8 * hi) - kcol))
          : NEG_INF;
      const float bm   = halfmax(s);
      const float mnew = fmaxf(m_[r], bm);
      if (mnew > -1e30f) {                       // uniform per half-wave
        const float alpha = (m_[r] > -1e30f) ? __expf(m_[r] - mnew) : 0.0f;
        const float e     = (s > -1e30f) ? __expf(s - mnew) : 0.0f;
        l_[r] = l_[r] * alpha + halfsum(e);
        m_[r] = mnew;
      }
    }
  }

  float rinv[8];
#pragma unroll
  for (int r = 0; r < 8; ++r) rinv[r] = (l_[r] > 0.0f) ? 1.0f / l_[r] : 0.0f;

  // ---- pass 2: probs -> LDS mean (ds_add_f32) + P@V via WMMA ----
  v8f acc[4] = {};
  for (int kc2 = 0; kc2 < 32; ++kc2) {
#pragma unroll
    for (int nb = 0; nb < 2; ++nb) {
      const int kc = kc2 * 2 + nb;
      const __bf16* kptr =
          kb + (size_t)((b << 10) + (kc << 4) + nl) * 1024 + (h << 6);
      const v8f c = score_tile(kptr, hi, a0, a1);
      const bool keep = kpm[(b << 10) + (kc << 4) + nl] != 0;
      const int kcol = (kc << 4) + nl;
#pragma unroll
      for (int r = 0; r < 8; ++r) {
        const float s = keep
            ? (c[r] * 0.125f + (float)((qb0 + r + 8 * hi) - kcol))
            : NEG_INF;
        const float p = (s > -1e30f) ? __expf(s - m_[r]) * rinv[r] : 0.0f;
        atomicAdd(&smean[(r + 8 * hi) * 1024 + kcol], p * 0.0625f);
        stg[(r + 8 * hi) * 32 + nb * 16 + nl] = f2bf(p);
      }
    }
    asm volatile("s_wait_dscnt 0" ::: "memory");   // LDS RAW within wave
    const v16bf pa = cat8(ld8(stg + nl * 32 + 16 * hi),
                          ld8(stg + nl * 32 + 8 + 16 * hi));
#pragma unroll
    for (int t = 0; t < 4; ++t) {
      const __bf16* vp = vtb +
          (size_t)(((b << 4) + h) * 64 + t * 16 + nl) * 1024 +
          kc2 * 32 + 16 * hi;
      acc[t] = wmma_bf16(pa, cat8(ld8(vp), ld8(vp + 8)), acc[t]);
    }
  }

  // ctx (already normalized) -> bf16
#pragma unroll
  for (int t = 0; t < 4; ++t)
#pragma unroll
    for (int r = 0; r < 8; ++r)
      ctxb[(size_t)((b << 10) + qb0 + r + 8 * hi) * 1024 +
           (h << 6) + t * 16 + nl] = f2bf(acc[t][r]);

  __syncthreads();
  // head-mean -> global attn_weights (contiguous 16 rows * 1024)
  float* dst = attn_out + (size_t)b * 1024 * 1024 + (size_t)qb0 * 1024;
  for (int i = threadIdx.x; i < 16 * 1024; i += 512) dst[i] = smean[i];
}

// ---------------------------------------------------------------------------
// Kernel 3: out projection.  out = ctx(bf16) * Wo(bf16)^T + bo   (fp32 out)
// One wave computes a 16x128 output tile.
// ---------------------------------------------------------------------------
__global__ void fmha_out_proj(const __bf16* __restrict__ ctxb,
                              const __bf16* __restrict__ wob,
                              const float* __restrict__ b_out,
                              float* __restrict__ out) {
  const int wid  = threadIdx.x >> 5;
  const int lane = threadIdx.x & 31;
  const int hi   = lane >> 4;
  const int nl   = lane & 15;
  const int tile = blockIdx.x * 8 + wid;     // 0..2047
  const int rt   = tile >> 3;
  const int ct   = tile & 7;
  const int row  = rt * 16 + nl;

  v8f acc[8] = {};
  for (int k0 = 0; k0 < 1024; k0 += 32) {
    const __bf16* ap = ctxb + (size_t)row * 1024 + k0;
    v16bf a = cat8(ld8(ap + 16 * hi), ld8(ap + 8 + 16 * hi));
#pragma unroll
    for (int t = 0; t < 8; ++t) {
      const int n = ct * 128 + t * 16 + nl;
      const __bf16* bp = wob + (size_t)n * 1024 + k0 + 16 * hi;
      acc[t] = wmma_bf16(a, cat8(ld8(bp), ld8(bp + 8)), acc[t]);
    }
  }
#pragma unroll
  for (int t = 0; t < 8; ++t) {
    const int col = ct * 128 + t * 16 + nl;
    const float bias = b_out[col];
#pragma unroll
    for (int r = 0; r < 8; ++r) {
      const int grow = rt * 16 + r + 8 * hi;
      out[(size_t)grow * 1024 + col] = acc[t][r] + bias;
    }
  }
}

// ---------------------------------------------------------------------------
extern "C" void kernel_launch(void* const* d_in, const int* in_sizes, int n_in,
                              void* d_out, int out_size, void* d_ws,
                              size_t ws_size, hipStream_t stream) {
  (void)in_sizes; (void)n_in; (void)out_size; (void)ws_size;
  const float* query  = (const float*)d_in[0];
  const float* key_   = (const float*)d_in[1];
  const float* value  = (const float*)d_in[2];
  const unsigned char* kpm = (const unsigned char*)d_in[3];
  const float* w_in   = (const float*)d_in[4];
  const float* b_in   = (const float*)d_in[5];
  const float* w_out  = (const float*)d_in[6];
  const float* b_out  = (const float*)d_in[7];
  float* out = (float*)d_out;

  const size_t N = 4ull * 1024 * 1024;   // 4194304 elements (B*S*E)
  __bf16* qb   = (__bf16*)d_ws;          // GEMM outputs
  __bf16* kb   = qb + N;
  __bf16* vtb  = kb + N;
  __bf16* ctxb = vtb + N;
  __bf16* xq   = ctxb + N;               // bf16 copies of inputs/weights
  __bf16* xk   = xq + N;
  __bf16* xv   = xk + N;
  __bf16* wib  = xv + N;                 // 3M elements
  __bf16* wob  = wib + 3ull * 1024 * 1024;  // 1M elements (total 64 MB ws)
  float* attn_out = out + N;             // second tuple element

  // one-shot fp32 -> bf16 conversions (2048 elems / block)
  fmha_cvt_bf16<<<dim3(2048), dim3(256), 0, stream>>>(query, xq);
  fmha_cvt_bf16<<<dim3(2048), dim3(256), 0, stream>>>(key_, xk);
  fmha_cvt_bf16<<<dim3(2048), dim3(256), 0, stream>>>(value, xv);
  fmha_cvt_bf16<<<dim3(1536), dim3(256), 0, stream>>>(w_in, wib);
  fmha_cvt_bf16<<<dim3(512),  dim3(256), 0, stream>>>(w_out, wob);

  fmha_qkv_proj<<<dim3(256, 3), dim3(256), 0, stream>>>(
      xq, xk, xv, wib, b_in, qb, kb, vtb);
  fmha_attention<<<dim3(64, 4), dim3(512), 81920, stream>>>(
      qb, kb, vtb, kpm, ctxb, attn_out);
  fmha_out_proj<<<dim3(256), dim3(256), 0, stream>>>(
      ctxb, wob, b_out, out);
}
